// GRAPH_CLASSIFIER_PLAIN_74105365725525
// MI455X (gfx1250) — matrix operand order repaired and checked by reading, not compiled, not run on hardware
//
#include <hip/hip_runtime.h>

// ---------------- static problem sizes (match reference) ----------------
#define BB    128
#define NNODE 512
#define EGQ   8192
#define FIN   128
#define HF    64
#define NC    8
#define EE    (BB * EGQ)   // 1048576 edges
#define NTOT  (BB * NNODE) // 65536 nodes
#define KP1   (NNODE / 2)  // 256 kept per graph after pool1
#define KP2   (KP1 / 2)    // 128 kept per graph after pool2
#define EPSF  1e-5f

#define CDIV(a, b) (((a) + (b) - 1) / (b))

typedef __attribute__((ext_vector_type(16))) __bf16 v16bf;
typedef __attribute__((ext_vector_type(8)))  float  v8f;

static __device__ __forceinline__ __bf16 f2bf(float x) {
    unsigned u = __builtin_bit_cast(unsigned, x);
    unsigned r = (u + 0x7FFFu + ((u >> 16) & 1u)) >> 16;   // round-to-nearest-even
    unsigned short hs = (unsigned short)r;
    return __builtin_bit_cast(__bf16, hs);
}

// ---------------------------------------------------------------------------
// WMMA GEMM: C[M,64] = A[M,K] * W[K,64] (+bias) (opt relu). K in {64,128},
// M multiple of 16 (grids chosen so all waves are full tiles).
// Weights are staged once per block into LDS, pre-converted to bf16 and
// pre-swizzled into the WMMA B-fragment layout, so each lane reads its
// 32-byte fragment with LDS b128 loads. One wave -> 16x64 output stripe.
// ---------------------------------------------------------------------------
template <int K>
__global__ void gemm_wmma_n64(const float* __restrict__ A,
                              const float* __restrict__ W,
                              const float* __restrict__ bias,
                              float* __restrict__ C,
                              int M, int relu) {
    constexpr int KS = K / 32;
    __shared__ __bf16 ldsb[K * 64];   // [KS][4 ntiles][32 lanes][16 elems]

    // cooperative stage: W[k, col] -> B-fragment element j of lane, ntile, kstep
    //   k = ks*32 + hl*16 + j ; col = nt*16 + (lane & 15) ; lane = hl*16 + lc
    for (int idx = threadIdx.x; idx < K * 64; idx += blockDim.x) {
        int k   = idx >> 6;
        int col = idx & 63;
        int ks  = k >> 5;
        int hl2 = (k >> 4) & 1;
        int j   = k & 15;
        int nt  = col >> 4;
        int ln  = hl2 * 16 + (col & 15);
        ldsb[(((ks * 4 + nt) * 32 + ln) << 4) + j] = f2bf(W[idx]);
    }
    __syncthreads();

    const int wave  = threadIdx.x >> 5;
    const int lane  = threadIdx.x & 31;
    const int tileM = (blockIdx.x * (blockDim.x >> 5) + wave) * 16;
    if (tileM >= M) return;            // wave-uniform: EXEC stays all-1s

    const int hl = lane >> 4;
    const int r  = lane & 15;

    v8f acc[4] = {};
    const float* __restrict__ arow = A + (size_t)(tileM + r) * K;

#pragma unroll
    for (int ks = 0; ks < KS; ++ks) {
        const int k0 = ks * 32;
        if (k0 + 32 < K) __builtin_prefetch(arow + k0 + 32, 0, 0);

        // ---- A fragment: contiguous k order, element j -> k = hl*16 + j (matches B staging) ----
        v16bf af;
#pragma unroll
        for (int v = 0; v < 8; ++v) {
            int kb = k0 + hl * 16 + 2 * v;
            float2 p = *(const float2*)(arow + kb);
            af[2 * v]     = f2bf(p.x);
            af[2 * v + 1] = f2bf(p.y);
        }
#pragma unroll
        for (int nt = 0; nt < 4; ++nt) {
            const v16bf bfv = *(const v16bf*)&ldsb[((ks * 4 + nt) * 32 + lane) << 4];
            acc[nt] = __builtin_amdgcn_wmma_f32_16x16x32_bf16(
                false, af, false, bfv, (short)0, acc[nt], false, false);
        }
    }

    // ---- store: 32-bit C/D 16x16 layout: VGPR i -> row i (lanes<16) / i+8 ----
#pragma unroll
    for (int nt = 0; nt < 4; ++nt) {
        int col = nt * 16 + r;
        float bv = bias ? bias[col] : 0.0f;
#pragma unroll
        for (int i = 0; i < 8; ++i) {
            int row = tileM + i + hl * 8;
            float v = acc[nt][i] + bv;
            if (relu) v = fmaxf(v, 0.0f);
            C[(size_t)row * 64 + col] = v;
        }
    }
}

// ---------------------------------------------------------------------------
// BatchNorm: column mean/var over n rows (64 features)
// ---------------------------------------------------------------------------
__global__ void bn_reduce(const float* __restrict__ x, float* __restrict__ mv, int n) {
    const int f = blockIdx.x;          // 0..63
    float s = 0.0f, q = 0.0f;
    for (int i = threadIdx.x; i < n; i += 256) {
        float v = x[(size_t)i * 64 + f];
        s += v; q += v * v;
    }
    __shared__ float ss[256], qq[256];
    ss[threadIdx.x] = s; qq[threadIdx.x] = q;
    __syncthreads();
    for (int st = 128; st > 0; st >>= 1) {
        if (threadIdx.x < st) {
            ss[threadIdx.x] += ss[threadIdx.x + st];
            qq[threadIdx.x] += qq[threadIdx.x + st];
        }
        __syncthreads();
    }
    if (threadIdx.x == 0) {
        float m = ss[0] / (float)n;
        mv[f]      = m;
        mv[64 + f] = qq[0] / (float)n - m * m;
    }
}

__global__ void bn_apply(float* __restrict__ x, const float* __restrict__ mv,
                         const float* __restrict__ g, const float* __restrict__ b, int n) {
    size_t idx = (size_t)blockIdx.x * blockDim.x + threadIdx.x;
    if (idx >= (size_t)n * 64) return;
    int f = (int)(idx & 63);
    x[idx] = g[f] * (x[idx] - mv[f]) * rsqrtf(mv[64 + f] + EPSF) + b[f];
}

// ---------------------------------------------------------------------------
// GCN support kernels
// ---------------------------------------------------------------------------
__global__ void fill_f32(float* __restrict__ p, float v, size_t n) {
    size_t i = (size_t)blockIdx.x * blockDim.x + threadIdx.x;
    if (i < n) p[i] = v;
}

__global__ void deg_accum(const int* __restrict__ dst, const float* __restrict__ em,
                          float* __restrict__ deg, int nE) {
    int e = blockIdx.x * blockDim.x + threadIdx.x;
    if (e >= nE) return;
    float m = em ? em[e] : 1.0f;
    if (m != 0.0f) atomicAdd(&deg[dst[e]], m);
}

__global__ void gcn_agg(const float* __restrict__ hw,
                        const int* __restrict__ src, const int* __restrict__ dst,
                        const float* __restrict__ em, const float* __restrict__ deg,
                        float* __restrict__ agg, int nE) {
    size_t idx = (size_t)blockIdx.x * blockDim.x + threadIdx.x;
    if (idx >= (size_t)nE * 64) return;
    int e = (int)(idx >> 6);
    int f = (int)(idx & 63);
    float m = em ? em[e] : 1.0f;
    if (m == 0.0f) return;
    int s = src[e], d = dst[e];
    float coef = m * rsqrtf(deg[s] * deg[d]);
    atomicAdd(&agg[(size_t)d * 64 + f], hw[(size_t)s * 64 + f] * coef);
}

__global__ void gcn_finalize(const float* __restrict__ hw, const float* __restrict__ agg,
                             const float* __restrict__ deg, const float* __restrict__ bias,
                             float* __restrict__ out, int n, int relu) {
    size_t idx = (size_t)blockIdx.x * blockDim.x + threadIdx.x;
    if (idx >= (size_t)n * 64) return;
    int i = (int)(idx >> 6);
    int f = (int)(idx & 63);
    float v = agg[idx] + hw[idx] / deg[i] + bias[f];
    if (relu) v = fmaxf(v, 0.0f);
    out[idx] = v;
}

// ---------------------------------------------------------------------------
// TopK pooling
// ---------------------------------------------------------------------------
__global__ void pool_score(const float* __restrict__ h, const float* __restrict__ w,
                           float* __restrict__ s, int n) {
    int i = blockIdx.x * blockDim.x + threadIdx.x;
    if (i >= n) return;
    float nrm = 0.0f;
#pragma unroll
    for (int f = 0; f < 64; ++f) nrm += w[f] * w[f];
    float d = 0.0f;
#pragma unroll
    for (int f = 0; f < 64; ++f) d += h[(size_t)i * 64 + f] * w[f];
    s[i] = fmaxf(d * rsqrtf(nrm), 0.0f);
}

// one block per graph; blockDim == nper (power of 2, <= 512).
// bitonic sort descending by score, ties -> lower original index first
// (matches jax.lax.top_k). Writes nodemap for every node and gathers new_x.
__global__ void topk_pool(const float* __restrict__ h, const float* __restrict__ s,
                          float* __restrict__ newx, int* __restrict__ nodemap,
                          int nper, int Kk) {
    __shared__ float key[512];
    __shared__ int   id[512];
    const int g = blockIdx.x;
    const int t = threadIdx.x;
    const int base = g * nper;

    key[t] = s[base + t];
    id[t]  = t;
    __syncthreads();

    for (int k = 2; k <= nper; k <<= 1) {
        for (int j = k >> 1; j > 0; j >>= 1) {
            int ixj = t ^ j;
            if (ixj > t) {
                float ka = key[t], kb = key[ixj];
                int   ia = id[t],  ib = id[ixj];
                bool aBefore = (ka > kb) || (ka == kb && ia < ib);
                bool desc = ((t & k) == 0);
                if (desc ? !aBefore : aBefore) {
                    key[t] = kb; key[ixj] = ka;
                    id[t]  = ib; id[ixj]  = ia;
                }
            }
            __syncthreads();
        }
    }

    int orig = id[t];
    int gold = base + orig;
    if (t < Kk) {
        int nid = g * Kk + t;
        nodemap[gold] = nid;
        float sc = key[t];
        for (int f = 0; f < 64; ++f)
            newx[(size_t)nid * 64 + f] = h[(size_t)gold * 64 + f] * sc;
    } else {
        nodemap[gold] = -1;
    }
}

__global__ void edge_remap(const int* __restrict__ si, const int* __restrict__ di,
                           const float* __restrict__ emi, const int* __restrict__ nodemap,
                           int* __restrict__ so, int* __restrict__ do_,
                           float* __restrict__ emo, int nE) {
    int e = blockIdx.x * blockDim.x + threadIdx.x;
    if (e >= nE) return;
    int a = nodemap[si[e]];
    int b = nodemap[di[e]];
    float m = emi ? emi[e] : 1.0f;
    bool val = (a >= 0) && (b >= 0);
    so[e]  = val ? a : 0;
    do_[e] = val ? b : 0;
    emo[e] = val ? m : 0.0f;
}

// ---------------------------------------------------------------------------
// readout + classifier head
// ---------------------------------------------------------------------------
__global__ void readout_mean(const float* __restrict__ h, float* __restrict__ gm) {
    int g = blockIdx.x;    // graph
    int f = threadIdx.x;   // feature
    float s = 0.0f;
    for (int j = 0; j < KP2; ++j) s += h[(size_t)(g * KP2 + j) * 64 + f];
    gm[g * 64 + f] = s / (float)KP2;
}

// block = 64 threads, one block per graph row. BN(t1)+relu, z@W2+b2, log_softmax.
__global__ void head_final(const float* __restrict__ t1, const float* __restrict__ mv,
                           const float* __restrict__ g5, const float* __restrict__ be5,
                           const float* __restrict__ W2, const float* __restrict__ b2,
                           float* __restrict__ out, int out_size) {
    __shared__ float z[64];
    __shared__ float lg[NC];
    const int rrow = blockIdx.x;
    const int f = threadIdx.x;
    float x = t1[rrow * 64 + f];
    z[f] = fmaxf(g5[f] * (x - mv[f]) * rsqrtf(mv[64 + f] + EPSF) + be5[f], 0.0f);
    __syncthreads();
    if (f < NC) {
        float acc = b2[f];
        for (int k = 0; k < 64; ++k) acc += z[k] * W2[k * NC + f];
        lg[f] = acc;
    }
    __syncthreads();
    if (f == 0) {
        float mx = lg[0];
        for (int c = 1; c < NC; ++c) mx = fmaxf(mx, lg[c]);
        float se = 0.0f;
        for (int c = 0; c < NC; ++c) se += expf(lg[c] - mx);
        float lse = mx + logf(se);
        for (int c = 0; c < NC; ++c) out[rrow * NC + c] = lg[c] - lse;
        if (rrow == 0 && out_size > BB * NC) out[BB * NC] = 0.0f;  // trailing scalar 0.0
    }
}

// ---------------------------------------------------------------------------
// workspace layout (4-byte elements)
// ---------------------------------------------------------------------------
static const size_t OFF_H   = 0;                               // NTOT*64
static const size_t OFF_W   = OFF_H   + (size_t)NTOT * 64;     // NTOT*64
static const size_t OFF_AGG = OFF_W   + (size_t)NTOT * 64;     // NTOT*64
static const size_t OFF_PX  = OFF_AGG + (size_t)NTOT * 64;     // BB*KP1*64
static const size_t OFF_DEG = OFF_PX  + (size_t)BB * KP1 * 64; // NTOT
static const size_t OFF_S   = OFF_DEG + NTOT;                  // NTOT
static const size_t OFF_MAP = OFF_S   + NTOT;                  // NTOT (int)
static const size_t OFF_E1S = OFF_MAP + NTOT;                  // EE (int)
static const size_t OFF_E1D = OFF_E1S + EE;                    // EE (int)
static const size_t OFF_E1M = OFF_E1D + EE;                    // EE (float)
static const size_t OFF_MV  = OFF_E1M + EE;                    // 128
static const size_t OFF_G   = OFF_MV  + 128;                   // BB*64
static const size_t OFF_T   = OFF_G   + (size_t)BB * 64;       // BB*64

extern "C" void kernel_launch(void* const* d_in, const int* in_sizes, int n_in,
                              void* d_out, int out_size, void* d_ws, size_t ws_size,
                              hipStream_t stream) {
    (void)in_sizes; (void)n_in; (void)ws_size;

    const float* x    = (const float*)d_in[0];
    const int*   ei   = (const int*)d_in[1];
    const int*   src0 = ei;
    const int*   dst0 = ei + EE;
    // d_in[2] = batch (unused; graphs are equal-sized)
    const float* Wpre = (const float*)d_in[3];
    const float* bpre = (const float*)d_in[4];
    const float* g0   = (const float*)d_in[5];
    const float* be0  = (const float*)d_in[6];
    const float* Wc1  = (const float*)d_in[7];
    const float* bc1  = (const float*)d_in[8];
    const float* g1   = (const float*)d_in[9];
    const float* be1  = (const float*)d_in[10];
    const float* pw1  = (const float*)d_in[11];
    const float* g2   = (const float*)d_in[12];
    const float* be2  = (const float*)d_in[13];
    const float* Wc2  = (const float*)d_in[14];
    const float* bc2  = (const float*)d_in[15];
    const float* g3   = (const float*)d_in[16];
    const float* be3  = (const float*)d_in[17];
    const float* pw2  = (const float*)d_in[18];
    const float* g4   = (const float*)d_in[19];
    const float* be4  = (const float*)d_in[20];
    const float* W1   = (const float*)d_in[21];
    const float* b1   = (const float*)d_in[22];
    const float* g5   = (const float*)d_in[23];
    const float* be5  = (const float*)d_in[24];
    const float* W2   = (const float*)d_in[25];
    const float* b2   = (const float*)d_in[26];

    float* ws   = (float*)d_ws;
    float* bufH = ws + OFF_H;
    float* bufW = ws + OFF_W;
    float* bufA = ws + OFF_AGG;
    float* bufP = ws + OFF_PX;
    float* deg  = ws + OFF_DEG;
    float* sc   = ws + OFF_S;
    int*   nmap = (int*)(ws + OFF_MAP);
    int*   e1s  = (int*)(ws + OFF_E1S);
    int*   e1d  = (int*)(ws + OFF_E1D);
    float* e1m  = ws + OFF_E1M;
    float* mv   = ws + OFF_MV;
    float* gbuf = ws + OFF_G;
    float* tbuf = ws + OFF_T;
    float* out  = (float*)d_out;

    const int TB = 256;
    const int GB = 256;     // gemm block: 8 waves, 8 M-tiles per block

    // ---- stage A: h = x @ W_pre + b_pre ; BN(g0,be0) ----
    gemm_wmma_n64<FIN><<<NTOT / 128, GB, 0, stream>>>(x, Wpre, bpre, bufH, NTOT, 0);
    bn_reduce<<<64, 256, 0, stream>>>(bufH, mv, NTOT);
    bn_apply<<<CDIV((size_t)NTOT * 64, TB), TB, 0, stream>>>(bufH, mv, g0, be0, NTOT);

    // ---- stage B: conv1 (GCN) + relu ; BN(g1,be1) ----
    gemm_wmma_n64<HF><<<NTOT / 128, GB, 0, stream>>>(bufH, Wc1, nullptr, bufW, NTOT, 0);
    fill_f32<<<CDIV((size_t)NTOT, TB), TB, 0, stream>>>(deg, 1.0f, NTOT);
    deg_accum<<<CDIV(EE, TB), TB, 0, stream>>>(dst0, nullptr, deg, EE);
    fill_f32<<<CDIV((size_t)NTOT * 64, TB), TB, 0, stream>>>(bufA, 0.0f, (size_t)NTOT * 64);
    gcn_agg<<<CDIV((size_t)EE * 64, TB), TB, 0, stream>>>(bufW, src0, dst0, nullptr, deg, bufA, EE);
    gcn_finalize<<<CDIV((size_t)NTOT * 64, TB), TB, 0, stream>>>(bufW, bufA, deg, bc1, bufH, NTOT, 1);
    bn_reduce<<<64, 256, 0, stream>>>(bufH, mv, NTOT);
    bn_apply<<<CDIV((size_t)NTOT * 64, TB), TB, 0, stream>>>(bufH, mv, g1, be1, NTOT);

    // ---- stage C: pool1 (top-256 of 512 per graph) ; BN(g2,be2) ----
    pool_score<<<CDIV(NTOT, TB), TB, 0, stream>>>(bufH, pw1, sc, NTOT);
    topk_pool<<<BB, NNODE, 0, stream>>>(bufH, sc, bufP, nmap, NNODE, KP1);
    edge_remap<<<CDIV(EE, TB), TB, 0, stream>>>(src0, dst0, nullptr, nmap, e1s, e1d, e1m, EE);
    const int n1 = BB * KP1;  // 32768
    bn_reduce<<<64, 256, 0, stream>>>(bufP, mv, n1);
    bn_apply<<<CDIV((size_t)n1 * 64, TB), TB, 0, stream>>>(bufP, mv, g2, be2, n1);

    // ---- stage D: conv2 (GCN) + relu ; BN(g3,be3) ----
    gemm_wmma_n64<HF><<<n1 / 128, GB, 0, stream>>>(bufP, Wc2, nullptr, bufW, n1, 0);
    fill_f32<<<CDIV((size_t)n1, TB), TB, 0, stream>>>(deg, 1.0f, n1);
    deg_accum<<<CDIV(EE, TB), TB, 0, stream>>>(e1d, e1m, deg, EE);
    fill_f32<<<CDIV((size_t)n1 * 64, TB), TB, 0, stream>>>(bufA, 0.0f, (size_t)n1 * 64);
    gcn_agg<<<CDIV((size_t)EE * 64, TB), TB, 0, stream>>>(bufW, e1s, e1d, e1m, deg, bufA, EE);
    gcn_finalize<<<CDIV((size_t)n1 * 64, TB), TB, 0, stream>>>(bufW, bufA, deg, bc2, bufP, n1, 1);
    bn_reduce<<<64, 256, 0, stream>>>(bufP, mv, n1);
    bn_apply<<<CDIV((size_t)n1 * 64, TB), TB, 0, stream>>>(bufP, mv, g3, be3, n1);

    // ---- stage E: pool2 (top-128 of 256 per graph) ; BN(g4,be4) ----
    // (pooled edges are never consumed by the output -> skip edge remap)
    pool_score<<<CDIV(n1, TB), TB, 0, stream>>>(bufP, pw2, sc, n1);
    topk_pool<<<BB, KP1, 0, stream>>>(bufP, sc, bufH, nmap, KP1, KP2);
    const int n2 = BB * KP2;  // 16384
    bn_reduce<<<64, 256, 0, stream>>>(bufH, mv, n2);
    bn_apply<<<CDIV((size_t)n2 * 64, TB), TB, 0, stream>>>(bufH, mv, g4, be4, n2);

    // ---- stage F: readout + head ----
    readout_mean<<<BB, 64, 0, stream>>>(bufH, gbuf);
    gemm_wmma_n64<HF><<<1, GB, 0, stream>>>(gbuf, W1, b1, tbuf, BB, 0);
    bn_reduce<<<64, 256, 0, stream>>>(tbuf, mv, BB);
    head_final<<<BB, 64, 0, stream>>>(tbuf, mv, g5, be5, W2, b2, out, out_size);
}
